// RingAttention_76364518523007
// MI455X (gfx1250) — matrix operand order repaired and checked by reading, not compiled, not run on hardware
//
#include <hip/hip_runtime.h>
#include <hip/hip_bf16.h>
#include <math.h>

typedef __bf16 bf16;
typedef __attribute__((ext_vector_type(16))) __bf16 v16bf;
typedef __attribute__((ext_vector_type(8)))  __bf16 v8bf;
typedef __attribute__((ext_vector_type(8)))  float  v8f;

#define N_SEQ   4096
#define DMODEL  1024
#define HEADS   16
#define DHEAD   64
#define NQKV    3072   // 3 * HEADS * DHEAD

// ---------------------------------------------------------------------------
// Fragment loaders.
// 16-bit A-matrix 16x32 (ISA 7.12.2): lane&15 = row M; lanes 0-15 hold
// K = {0..7, 16..23}, lanes 16-31 hold K = {8..15, 24..31}. Element i<8 ->
// K=base0+i, i>=8 -> K=base1+(i-8). Two 16-byte loads.
// 16-bit B-matrix 32x16: lane&15 = col N; same K split as A (lanes 0-15 hold
// K = {0..7, 16..23}, lanes 16-31 hold K = {8..15, 24..31}); two 16-byte loads.
// ---------------------------------------------------------------------------
__device__ __forceinline__ v16bf frag_split(const bf16* p0, const bf16* p1) {
  v8bf lo = *(const v8bf*)p0;
  v8bf hi = *(const v8bf*)p1;
  v16bf r;
#pragma unroll
  for (int i = 0; i < 8; ++i) { r[i] = lo[i]; r[i + 8] = hi[i]; }
  return r;
}
__device__ __forceinline__ v16bf frag_contig(const bf16* p) {
  return frag_split(p, p + 8);
}

// ---------------------------------------------------------------------------
// Weight convert + transpose: wT[n][k] = (bf16) w[k][n]
// ---------------------------------------------------------------------------
__global__ __launch_bounds__(256) void cvt_wqkvT_kernel(const float* __restrict__ w,
                                                        bf16* __restrict__ wT) {
  int idx = blockIdx.x * 256 + threadIdx.x;        // idx = n*1024 + k, n < 3072
  int n = idx >> 10, kk = idx & 1023;
  wT[idx] = (bf16)w[(size_t)kk * NQKV + n];
}
__global__ __launch_bounds__(256) void cvt_woutT_kernel(const float* __restrict__ w,
                                                        bf16* __restrict__ wT) {
  int idx = blockIdx.x * 256 + threadIdx.x;        // idx = n*1024 + k, n < 1024
  int n = idx >> 10, kk = idx & 1023;
  wT[idx] = (bf16)w[(size_t)kk * DMODEL + n];
}

// ---------------------------------------------------------------------------
// RMSNorm (reference: x / max(||x||2, 1e-12) * sqrt(1024) * gamma), bf16 out.
// ---------------------------------------------------------------------------
__global__ __launch_bounds__(256) void rmsnorm_kernel(const float* __restrict__ x,
                                                      const float* __restrict__ gamma,
                                                      bf16* __restrict__ xn) {
  __shared__ float red[8];
  const int row = blockIdx.x;
  const float* xr = x + (size_t)row * DMODEL;
  float s = 0.f;
  for (int i = threadIdx.x; i < DMODEL; i += 256) { float v = xr[i]; s += v * v; }
#pragma unroll
  for (int off = 16; off > 0; off >>= 1) s += __shfl_xor(s, off, 32);
  if ((threadIdx.x & 31) == 0) red[threadIdx.x >> 5] = s;
  __syncthreads();
  if (threadIdx.x < 8) {
    float t = red[threadIdx.x];
#pragma unroll
    for (int off = 4; off > 0; off >>= 1) t += __shfl_xor(t, off, 32);
    if (threadIdx.x == 0) red[0] = t;
  }
  __syncthreads();
  const float l2 = fmaxf(sqrtf(red[0]), 1e-12f);
  const float scale = 32.0f / l2;                  // sqrt(1024) = 32
  for (int i = threadIdx.x; i < DMODEL; i += 256)
    xn[(size_t)row * DMODEL + i] = (bf16)(xr[i] * scale * gamma[i]);
}

// ---------------------------------------------------------------------------
// QKV GEMM: [4096 x 1024] x [1024 x 3072] -> routed q / k / vT buffers.
// One wave per 64x64 output tile: 4 A-frags x 4 B-frags -> 16 WMMAs per
// K-step of 32. K-loop unrolled x2 so the scheduler can interleave two
// independent load/WMMA chains (kills WMMA->load WAR nops). Prefetch is
// unconditional: TH=0 prefetch is speculative and silently dropped if the
// address is invalid (ISA 10.5), so the end-of-row over-prefetch is safe.
// q is pre-scaled by sqrt(DHEAD)=8 (reference multiplies, not divides).
// v is written transposed [h][64][4096] for contiguous P.V B-fragments.
// ---------------------------------------------------------------------------
__global__ __launch_bounds__(32) void qkv_gemm_kernel(const bf16* __restrict__ xn,
                                                      const bf16* __restrict__ wT,
                                                      bf16* __restrict__ qb,
                                                      bf16* __restrict__ kb,
                                                      bf16* __restrict__ vT) {
  const int lane = threadIdx.x;
  const int l15  = lane & 15;
  const int hiH  = lane >> 4;
  const int tn   = blockIdx.x;   // 0..47
  const int tm   = blockIdx.y;   // 0..63

  const bf16* arow[4];
  const bf16* brow[4];
#pragma unroll
  for (int f = 0; f < 4; ++f) {
    arow[f] = xn + (size_t)(tm * 64 + f * 16 + l15) * DMODEL;
    brow[f] = wT + (size_t)(tn * 64 + f * 16 + l15) * DMODEL;
  }

  const int a0 = hiH ? 8 : 0, a1 = hiH ? 24 : 16;  // A-frag K offsets
  const int bb = hiH ? 16 : 0;                     // B prefetch K offset
  v8f acc[4][4] = {};

#pragma unroll 2
  for (int ks = 0; ks < DMODEL; ks += 32) {
    __builtin_prefetch(arow[0] + ks + 64 + a0, 0, 3);
    __builtin_prefetch(brow[0] + ks + 64 + bb, 0, 3);
    v16bf A[4], B[4];
#pragma unroll
    for (int f = 0; f < 4; ++f) {
      A[f] = frag_split(arow[f] + ks + a0, arow[f] + ks + a1);
      B[f] = frag_split(brow[f] + ks + a0, brow[f] + ks + a1);
    }
#pragma unroll
    for (int fm = 0; fm < 4; ++fm)
#pragma unroll
      for (int fn = 0; fn < 4; ++fn)
        acc[fm][fn] = __builtin_amdgcn_wmma_f32_16x16x32_bf16(
            false, A[fm], false, B[fn], (short)0, acc[fm][fn], false, false);
  }

  // C layout: VGPR r -> row (lane<16 ? r : r+8), col = lane&15.
#pragma unroll
  for (int fm = 0; fm < 4; ++fm)
#pragma unroll
    for (int fn = 0; fn < 4; ++fn)
#pragma unroll
      for (int r = 0; r < 8; ++r) {
        const int row = tm * 64 + fm * 16 + (hiH ? r + 8 : r);
        const int col = tn * 64 + fn * 16 + l15;
        const float val = acc[fm][fn][r];
        const int which = col >> 10;         // 0=q 1=k 2=v
        const int rem   = col & 1023;
        const int h  = rem >> 6;
        const int dd = rem & 63;
        if (which == 0)
          qb[((size_t)h * N_SEQ + row) * DHEAD + dd] = (bf16)(val * 8.0f);
        else if (which == 1)
          kb[((size_t)h * N_SEQ + row) * DHEAD + dd] = (bf16)val;
        else
          vT[((size_t)h * DHEAD + dd) * N_SEQ + row] = (bf16)val;
      }
}

// ---------------------------------------------------------------------------
// Causal flash attention: 1 wave per (16-query tile, head), 64 keys/iter.
// Per 64-key block: S = Q K^T (8 WMMAs), causal mask + online softmax via a
// 16x64 LDS tile (C-layout -> A-layout reshuffle, exp on re-read), O rescale,
// O += P V (8 WMMAs; vT gives contiguous B-frags over the key dimension).
// ---------------------------------------------------------------------------
__global__ __launch_bounds__(32) void attn_kernel(const bf16* __restrict__ qb,
                                                  const bf16* __restrict__ kb,
                                                  const bf16* __restrict__ vT,
                                                  bf16* __restrict__ ao) {
  __shared__ float sS[16][65];
  __shared__ float sM[16], sF[16], sI[16];

  const int lane = threadIdx.x;
  const int l15  = lane & 15;
  const int hiH  = lane >> 4;
  const int qt   = blockIdx.x;   // 0..255
  const int h    = blockIdx.y;   // 0..15

  const bf16* qh = qb + (size_t)h * N_SEQ * DHEAD;
  const bf16* kh = kb + (size_t)h * N_SEQ * DHEAD;
  const bf16* vh = vT + (size_t)h * DHEAD * N_SEQ;

  const int a0  = hiH ? 8 : 0, a1 = hiH ? 24 : 16;
  const int bko = hiH ? 16 : 0;

  // Q A-fragments for the two d-chunks (d = 0..31, 32..63), loaded once.
  v16bf aq0, aq1;
  {
    const bf16* qrow = qh + (size_t)(qt * 16 + l15) * DHEAD;
    aq0 = frag_split(qrow + a0,      qrow + a1);
    aq1 = frag_split(qrow + 32 + a0, qrow + 32 + a1);
  }

  v8f o[4] = {};                       // 16 rows x 64 dims, 4 dim tiles
  float m_old = -3.0e38f, lsum = 0.f;  // owned per-row by lanes 0..15

  const int nkb = ((qt * 16 + 15) >> 6) + 1;   // 64-key blocks (causal bound)
  for (int kblk = 0; kblk < nkb; ++kblk) {
    // ---- S = Q K^T for 64 keys (4 half tiles x 2 d-chunks) ----
    v8f s[4] = {};
#pragma unroll
    for (int t = 0; t < 4; ++t) {
      const bf16* krow = kh + (size_t)(kblk * 64 + t * 16 + l15) * DHEAD;
      v16bf bk0 = frag_split(krow + a0, krow + a1);
      v16bf bk1 = frag_split(krow + 32 + a0, krow + 32 + a1);
      s[t] = __builtin_amdgcn_wmma_f32_16x16x32_bf16(false, aq0, false, bk0, (short)0, s[t], false, false);
      s[t] = __builtin_amdgcn_wmma_f32_16x16x32_bf16(false, aq1, false, bk1, (short)0, s[t], false, false);
    }

    __syncthreads();                   // WAR guard on sS vs prior P read
    // ---- write S with causal mask (C-layout scatter) ----
#pragma unroll
    for (int t = 0; t < 4; ++t)
#pragma unroll
      for (int r = 0; r < 8; ++r) {
        const int mrow = hiH ? r + 8 : r;
        const int gi = qt * 16 + mrow;
        const int gj = kblk * 64 + t * 16 + l15;
        sS[mrow][t * 16 + l15] = (gj <= gi) ? s[t][r] : -3.0e38f;
      }
    __syncthreads();

    // ---- online softmax row stats (lane r owns row r) ----
    if (lane < 16) {
      float rmax = -3.0e38f;
#pragma unroll
      for (int c = 0; c < 64; ++c) rmax = fmaxf(rmax, sS[lane][c]);
      const float m_new = fmaxf(m_old, rmax);
      const float fac   = __expf(m_old - m_new);
      float se = 0.f;
#pragma unroll
      for (int c = 0; c < 64; ++c) se += __expf(sS[lane][c] - m_new);
      lsum = lsum * fac + se;
      m_old = m_new;
      sM[lane] = m_new;
      sF[lane] = fac;
    }
    __syncthreads();

    // ---- rescale O accumulators ----
#pragma unroll
    for (int dt = 0; dt < 4; ++dt)
#pragma unroll
      for (int r = 0; r < 8; ++r) o[dt][r] *= sF[hiH ? r + 8 : r];

    // ---- build P A-fragments (exp on LDS re-read, bf16 convert) ----
    v16bf p0, p1;                       // keys 0..31 and 32..63 of the block
    {
      const float mrow = sM[l15];
#pragma unroll
      for (int i = 0; i < 8; ++i) {
        p0[i]     = (bf16)__expf(sS[l15][a0 + i] - mrow);
        p0[i + 8] = (bf16)__expf(sS[l15][a1 + i] - mrow);
        p1[i]     = (bf16)__expf(sS[l15][32 + a0 + i] - mrow);
        p1[i + 8] = (bf16)__expf(sS[l15][32 + a1 + i] - mrow);
      }
    }

    // ---- O += P V ----
#pragma unroll
    for (int dt = 0; dt < 4; ++dt) {
      const bf16* vrow = vh + (size_t)(dt * 16 + l15) * N_SEQ + kblk * 64;
      v16bf bv0 = frag_split(vrow + a0, vrow + a1);
      v16bf bv1 = frag_split(vrow + 32 + a0, vrow + 32 + a1);
      o[dt] = __builtin_amdgcn_wmma_f32_16x16x32_bf16(false, p0, false, bv0, (short)0, o[dt], false, false);
      o[dt] = __builtin_amdgcn_wmma_f32_16x16x32_bf16(false, p1, false, bv1, (short)0, o[dt], false, false);
    }
  }

  if (lane < 16) sI[lane] = 1.0f / lsum;
  __syncthreads();

#pragma unroll
  for (int dt = 0; dt < 4; ++dt)
#pragma unroll
    for (int r = 0; r < 8; ++r) {
      const int mrow = hiH ? r + 8 : r;
      const int gi = qt * 16 + mrow;
      ao[(size_t)gi * DMODEL + h * DHEAD + dt * 16 + l15] = (bf16)(o[dt][r] * sI[mrow]);
    }
}

// ---------------------------------------------------------------------------
// Output projection: [4096 x 1024] x [1024 x 1024] -> fp32 d_out.
// Same 64x64-per-wave scheme as the QKV GEMM.
// ---------------------------------------------------------------------------
__global__ __launch_bounds__(32) void out_gemm_kernel(const bf16* __restrict__ ao,
                                                      const bf16* __restrict__ wT,
                                                      float* __restrict__ out) {
  const int lane = threadIdx.x;
  const int l15  = lane & 15;
  const int hiH  = lane >> 4;
  const int tn   = blockIdx.x;   // 0..15
  const int tm   = blockIdx.y;   // 0..63

  const bf16* arow[4];
  const bf16* brow[4];
#pragma unroll
  for (int f = 0; f < 4; ++f) {
    arow[f] = ao + (size_t)(tm * 64 + f * 16 + l15) * DMODEL;
    brow[f] = wT + (size_t)(tn * 64 + f * 16 + l15) * DMODEL;
  }

  const int a0 = hiH ? 8 : 0, a1 = hiH ? 24 : 16;
  const int bb = hiH ? 16 : 0;
  v8f acc[4][4] = {};

#pragma unroll 2
  for (int ks = 0; ks < DMODEL; ks += 32) {
    __builtin_prefetch(arow[0] + ks + 64 + a0, 0, 3);
    __builtin_prefetch(brow[0] + ks + 64 + bb, 0, 3);
    v16bf A[4], B[4];
#pragma unroll
    for (int f = 0; f < 4; ++f) {
      A[f] = frag_split(arow[f] + ks + a0, arow[f] + ks + a1);
      B[f] = frag_split(brow[f] + ks + a0, brow[f] + ks + a1);
    }
#pragma unroll
    for (int fm = 0; fm < 4; ++fm)
#pragma unroll
      for (int fn = 0; fn < 4; ++fn)
        acc[fm][fn] = __builtin_amdgcn_wmma_f32_16x16x32_bf16(
            false, A[fm], false, B[fn], (short)0, acc[fm][fn], false, false);
  }

#pragma unroll
  for (int fm = 0; fm < 4; ++fm)
#pragma unroll
    for (int fn = 0; fn < 4; ++fn)
#pragma unroll
      for (int r = 0; r < 8; ++r) {
        const int row = tm * 64 + fm * 16 + (hiH ? r + 8 : r);
        const int col = tn * 64 + fn * 16 + l15;
        out[(size_t)row * DMODEL + col] = acc[fm][fn][r];
      }
}

// ---------------------------------------------------------------------------
extern "C" void kernel_launch(void* const* d_in, const int* in_sizes, int n_in,
                              void* d_out, int out_size, void* d_ws, size_t ws_size,
                              hipStream_t stream) {
  (void)in_sizes; (void)n_in; (void)out_size; (void)ws_size;

  const float* x      = (const float*)d_in[0];   // [1,4096,1024]
  const float* gamma  = (const float*)d_in[1];   // [1024]
  const float* w_qkv  = (const float*)d_in[2];   // [1024,3072]
  const float* w_out  = (const float*)d_in[3];   // [1024,1024]
  float*       out    = (float*)d_out;           // [1,4096,1024]

  // Workspace carve-up (~40 MB). attn output reuses the xn buffer (xn is
  // dead after the QKV GEMM).
  char* ws = (char*)d_ws;
  bf16* xn    = (bf16*)ws;  ws += (size_t)N_SEQ * DMODEL * 2;   // 8 MB (also ao)
  bf16* wqkvT = (bf16*)ws;  ws += (size_t)NQKV * DMODEL * 2;    // 6 MB
  bf16* woutT = (bf16*)ws;  ws += (size_t)DMODEL * DMODEL * 2;  // 2 MB
  bf16* qb    = (bf16*)ws;  ws += (size_t)HEADS * N_SEQ * DHEAD * 2;  // 8 MB
  bf16* kb    = (bf16*)ws;  ws += (size_t)HEADS * N_SEQ * DHEAD * 2;  // 8 MB
  bf16* vT    = (bf16*)ws;  ws += (size_t)HEADS * DHEAD * N_SEQ * 2;  // 8 MB
  bf16* ao    = xn;

  cvt_wqkvT_kernel<<<(NQKV * DMODEL) / 256, 256, 0, stream>>>(w_qkv, wqkvT);
  cvt_woutT_kernel<<<(DMODEL * DMODEL) / 256, 256, 0, stream>>>(w_out, woutT);
  rmsnorm_kernel<<<N_SEQ, 256, 0, stream>>>(x, gamma, xn);
  qkv_gemm_kernel<<<dim3(NQKV / 64, N_SEQ / 64), 32, 0, stream>>>(xn, wqkvT, qb, kb, vT);
  attn_kernel<<<dim3(N_SEQ / 16, HEADS), 32, 0, stream>>>(qb, kb, vT, ao);
  out_gemm_kernel<<<dim3(DMODEL / 64, N_SEQ / 64), 32, 0, stream>>>(ao, woutT, out);
}
